// EquiformerV2_31997506355368
// MI455X (gfx1250) — compile-verified, matrix operand order checked by reading, not run on hardware
//
#include <hip/hip_runtime.h>

#define NATOM  2048
#define NEDGE  40960
#define CCH    128
#define NCOEF  25
#define ROWS_N (NATOM*NCOEF)   // 51200
#define ROWS_E (NEDGE*NCOEF)   // 1024000
#define NLAY   2
#define NHEAD  8
#define ACH    32
#define FFH    512
#define NBASIS 600
#define NBPAD  608             // 600 padded to multiple of 32
#define EPSF   1e-6f
#define DELTA  (12.0f/599.0f)
#define GCOEF  (-0.5f/((2.0f*DELTA)*(2.0f*DELTA)))

typedef __attribute__((ext_vector_type(16))) __bf16    bf16x16;
typedef __attribute__((ext_vector_type(8)))  float     f32x8;
typedef __attribute__((ext_vector_type(4)))  unsigned  u32x4;
typedef __attribute__((ext_vector_type(4)))  float     f32x4;

// ---------------- scalar helpers ----------------
__device__ __forceinline__ unsigned short f2bf(float f){
  unsigned u = __float_as_uint(f);
  unsigned r = u + 0x7FFFu + ((u >> 16) & 1u);
  return (unsigned short)(r >> 16);
}
__device__ __forceinline__ float bf2f(unsigned short b){
  return __uint_as_float(((unsigned)b) << 16);
}
__device__ __forceinline__ float silu_f(float x){ return x / (1.f + __expf(-x)); }

__device__ __forceinline__ unsigned f2ord(float f){
  unsigned u = __float_as_uint(f);
  return (u & 0x80000000u) ? ~u : (u | 0x80000000u);
}
__device__ __forceinline__ float ord2f(unsigned u){
  unsigned v = (u & 0x80000000u) ? (u & 0x7FFFFFFFu) : ~u;
  return __uint_as_float(v);
}

__device__ __forceinline__ f32x8 zero8(){
  f32x8 z;
#pragma unroll
  for (int i=0;i<8;++i) z[i]=0.f;
  return z;
}

__device__ __forceinline__ f32x8 wmma_bf(bf16x16 a, bf16x16 b, f32x8 c){
  return __builtin_amdgcn_wmma_f32_16x16x32_bf16(false, a, false, b, (short)0, c, false, false);
}

// k offset inside a 32-wide k-tile for fragment register p (0..7), half = lane>>4
__device__ __forceinline__ int kof(int p, int half){
  return ((p>>2)<<4) + (half<<3) + ((p&3)<<1);
}

// ---------------- fragment loaders ----------------
// A fragment from a row-major bf16 row: two contiguous 16-byte loads.
__device__ __forceinline__ bf16x16 loadA_row(const unsigned short* row, int kbase, int half){
  union { bf16x16 v; u32x4 q[2]; } f;
  f.q[0] = *(const u32x4*)(row + kbase + 8*half);
  f.q[1] = *(const u32x4*)(row + kbase + 16 + 8*half);
  return f.v;
}
// A fragment from a row-major f32 row (contiguous groups, converted).
__device__ __forceinline__ bf16x16 loadA_row_f32(const float* row, int kbase, int half){
  union { bf16x16 v; unsigned short u[16]; } f;
#pragma unroll
  for (int g=0; g<2; ++g){
    f32x4 a = *(const f32x4*)(row + kbase + 16*g + 8*half);
    f32x4 b = *(const f32x4*)(row + kbase + 16*g + 8*half + 4);
#pragma unroll
    for (int q=0;q<4;++q){ f.u[8*g+q] = f2bf(a[q]); f.u[8*g+4+q] = f2bf(b[q]); }
  }
  return f.v;
}
// B fragment from pre-packed weight: tiles of (kt, nt), 32 bytes per lane, contiguous.
__device__ __forceinline__ bf16x16 loadBp(const unsigned short* Wp, int ntiles_n, int kt, int nt, int lane){
  const unsigned* base = (const unsigned*)Wp + ((size_t)(kt*ntiles_n + nt)*32 + lane)*8;
  union { bf16x16 v; u32x4 q[2]; } f;
  f.q[0] = *(const u32x4*)base;
  f.q[1] = *(const u32x4*)(base + 4);
  return f.v;
}

// ---------------- pack [K,N] f32 weight into B-fragment layout (bf16, K padded to Kp) --------
__global__ void k_pack_b(const float* __restrict__ src, unsigned short* __restrict__ dst,
                         int K, int N, int Kp){
  int idx = blockIdx.x*blockDim.x + threadIdx.x;          // one dword per thread
  int total = (Kp>>5)*(N>>4)*256;
  if (idx >= total) return;
  int p    = idx & 7;
  int lane = (idx >> 3) & 31;
  int tile = idx >> 8;                                    // kt*(N/16)+nt
  int ntn  = N >> 4;
  int kt = tile / ntn, nt = tile - kt*ntn;
  int half = lane >> 4, r = lane & 15;
  int k = kt*32 + kof(p, half);
  int n = nt*16 + r;
  unsigned short lo = (k   < K) ? f2bf(src[(size_t)k*N + n])     : (unsigned short)0;
  unsigned short hi = (k+1 < K) ? f2bf(src[(size_t)(k+1)*N + n]) : (unsigned short)0;
  ((unsigned*)dst)[idx] = (unsigned)lo | ((unsigned)hi << 16);
}

// ---------------- utility kernels ----------------
__global__ void k_zero_f(float* p, int n){
  int i = blockIdx.x*blockDim.x + threadIdx.x;
  if (i < n) p[i] = 0.f;
}
__global__ void k_zero_u(unsigned* p, int n){
  int i = blockIdx.x*blockDim.x + threadIdx.x;
  if (i < n) p[i] = 0u;
}
__global__ void k_dist(const float* __restrict__ pos, const int* __restrict__ esrc,
                       const int* __restrict__ edst, float* __restrict__ dist){
  int e = blockIdx.x*blockDim.x + threadIdx.x;
  if (e < NEDGE){
    int j = esrc[e], i = edst[e];
    float dx = pos[j*3+0]-pos[i*3+0];
    float dy = pos[j*3+1]-pos[i*3+1];
    float dz = pos[j*3+2]-pos[i*3+2];
    dist[e] = sqrtf(dx*dx+dy*dy+dz*dz+EPSF);
  }
}
__global__ void k_init_x(const float* __restrict__ emb, const int* __restrict__ anum,
                         float* __restrict__ x){
  int idx = blockIdx.x*blockDim.x + threadIdx.x;
  if (idx < NATOM*NCOEF*CCH){
    int node = idx / (NCOEF*CCH);
    int rem  = idx - node*(NCOEF*CCH);
    int lc = rem >> 7, c = rem & 127;
    x[idx] = (lc == 0) ? emb[anum[node]*CCH + c] : 0.f;
  }
}

// ---------------- RMS norm (per node over NC*C) ----------------
__global__ void k_rms(const float* __restrict__ x, const float* __restrict__ s,
                      unsigned short* __restrict__ xn){
  __shared__ float red[256];
  int node = blockIdx.x, tid = threadIdx.x;
  const float* xr = x + (size_t)node*NCOEF*CCH;
  float ss = 0.f;
  for (int idx = tid; idx < NCOEF*CCH; idx += 256){ float v = xr[idx]; ss += v*v; }
  red[tid] = ss; __syncthreads();
  for (int st = 128; st > 0; st >>= 1){
    if (tid < st) red[tid] += red[tid+st];
    __syncthreads();
  }
  float scale = rsqrtf(red[0] / (float)(NCOEF*CCH) + EPSF);
  for (int idx = tid; idx < NCOEF*CCH; idx += 256)
    xn[(size_t)node*NCOEF*CCH + idx] = f2bf(xr[idx]*scale*s[idx & 127]);
}

// ---------------- radial MLP stage 1: e1 = silu(basis @ Wrad1 + b1) ----------------
__global__ void k_rad1(const float* __restrict__ dist,
                       const unsigned short* __restrict__ Wp,   // packed [608,128]
                       const float* __restrict__ bias,          // [128]
                       unsigned short* __restrict__ out){       // [E,128] bf16
  int wv = threadIdx.x >> 5, lane = threadIdx.x & 31;
  int half = lane >> 4, r = lane & 15;
  int e0 = (blockIdx.x*4 + wv) * 16;
  float d = dist[e0 + r];
  f32x8 acc[8];
#pragma unroll
  for (int nt=0;nt<8;++nt) acc[nt] = zero8();
  for (int kt = 0; kt < 19; ++kt){
    int kbase = kt*32;
    union { bf16x16 v; unsigned short u[16]; } af;
#pragma unroll
    for (int g=0; g<2; ++g){
#pragma unroll
      for (int q=0;q<8;++q){
        int k = kbase + 16*g + 8*half + q;
        float t = d - (float)k*DELTA;
        float b = __expf(GCOEF*t*t);
        af.u[8*g+q] = f2bf((k < NBASIS) ? b : 0.f);   // branchless select
      }
    }
#pragma unroll
    for (int nt=0;nt<8;++nt)
      acc[nt] = wmma_bf(af.v, loadBp(Wp, 8, kt, nt, lane), acc[nt]);
  }
#pragma unroll
  for (int nt=0;nt<8;++nt){
#pragma unroll
    for (int p=0;p<8;++p){
      int m = p + 8*half, col = nt*16 + r;
      float v = silu_f(acc[nt][p] + bias[col]);
      out[(size_t)(e0+m)*CCH + col] = f2bf(v);
    }
  }
}

// ---------------- generic [rows,128] @ [128,128] GEMM ----------------
__global__ void k_gemm128(const unsigned short* __restrict__ Abf, const float* __restrict__ A32,
                          const unsigned short* __restrict__ Wp, const float* __restrict__ bias,
                          unsigned short* __restrict__ outbf, float* __restrict__ out32,
                          int rows, int do_silu, int do_acc){
  int wv = threadIdx.x >> 5, lane = threadIdx.x & 31;
  int half = lane >> 4, r = lane & 15;
  int row0 = (blockIdx.x*4 + wv) * 16;
  if (row0 >= rows) return;
  bf16x16 a[4];
  if (A32){
    const float* rp = A32 + (size_t)(row0+r)*CCH;
#pragma unroll
    for (int kt=0;kt<4;++kt) a[kt] = loadA_row_f32(rp, kt*32, half);
  } else {
    const unsigned short* rp = Abf + (size_t)(row0+r)*CCH;
#pragma unroll
    for (int kt=0;kt<4;++kt) a[kt] = loadA_row(rp, kt*32, half);
  }
  f32x8 acc[8];
#pragma unroll
  for (int nt=0;nt<8;++nt) acc[nt] = zero8();
#pragma unroll
  for (int nt=0;nt<8;++nt)
#pragma unroll
    for (int kt=0;kt<4;++kt)
      acc[nt] = wmma_bf(a[kt], loadBp(Wp, 8, kt, nt, lane), acc[nt]);
#pragma unroll
  for (int nt=0;nt<8;++nt){
#pragma unroll
    for (int p=0;p<8;++p){
      int m = p + 8*half, col = nt*16 + r;
      float v = acc[nt][p];
      if (bias) v += bias[col];
      if (do_silu) v = silu_f(v);
      size_t o = (size_t)(row0+m)*CCH + col;
      if (outbf) outbf[o] = f2bf(v);
      if (out32){ if (do_acc) out32[o] += v; else out32[o] = v; }
    }
  }
}

// ---------------- attention logits: silu(concat(xt0,xs0)@W_alpha) . alpha_vec ----------------
__global__ void k_alpha(const unsigned short* __restrict__ xn,
                        const unsigned short* __restrict__ Wap,  // packed [256,256]
                        const float* __restrict__ avec,          // [8,32]
                        const int* __restrict__ esrc, const int* __restrict__ edst,
                        float* __restrict__ logits){
  __shared__ float lacc[4][16][8];
  int wv = threadIdx.x >> 5, lane = threadIdx.x & 31;
  int half = lane >> 4, r = lane & 15;
  int e0 = (blockIdx.x*4 + wv) * 16;
  for (int t = threadIdx.x; t < 4*16*8; t += blockDim.x) ((float*)lacc)[t] = 0.f;
  __syncthreads();
  int er = e0 + r;
  int jN = esrc[er], iN = edst[er];
  const unsigned short* rowT = xn + (size_t)iN*NCOEF*CCH;  // target (first 128 cols)
  const unsigned short* rowS = xn + (size_t)jN*NCOEF*CCH;  // source (last 128 cols)
  bf16x16 a[8];
#pragma unroll
  for (int kt=0;kt<4;++kt) a[kt]   = loadA_row(rowT, kt*32, half);
#pragma unroll
  for (int kt=0;kt<4;++kt) a[kt+4] = loadA_row(rowS, kt*32, half);
  for (int nt=0;nt<16;++nt){
    f32x8 acc = zero8();
#pragma unroll
    for (int kt=0;kt<8;++kt)
      acc = wmma_bf(a[kt], loadBp(Wap, 16, kt, nt, lane), acc);
    int h = nt >> 1;
    float vv = avec[h*ACH + ((nt&1)<<4) + r];
#pragma unroll
    for (int p=0;p<8;++p){
      int m = p + 8*half;
      atomicAdd(&lacc[wv][m][h], silu_f(acc[p]) * vv);
    }
  }
  __syncthreads();
  for (int t = lane; t < 16*8; t += 32){
    int m = t >> 3, h = t & 7;
    logits[(size_t)(e0+m)*NHEAD + h] = lacc[wv][m][h];
  }
}

// ---------------- segment softmax ----------------
__global__ void k_seg_max(const float* __restrict__ logits, const int* __restrict__ edst,
                          unsigned* __restrict__ nmax){
  int idx = blockIdx.x*blockDim.x + threadIdx.x;
  if (idx < NEDGE*NHEAD){
    int e = idx >> 3, h = idx & 7;
    atomicMax(&nmax[edst[e]*NHEAD + h], f2ord(logits[idx]));
  }
}
__global__ void k_seg_exp(const float* __restrict__ logits, const int* __restrict__ edst,
                          const unsigned* __restrict__ nmax, float* __restrict__ nsum,
                          float* __restrict__ attw){
  int idx = blockIdx.x*blockDim.x + threadIdx.x;
  if (idx < NEDGE*NHEAD){
    int e = idx >> 3, h = idx & 7;
    float m = ord2f(nmax[edst[e]*NHEAD + h]);
    float z = __expf(logits[idx] - m);
    attw[idx] = z;
    atomicAdd(&nsum[edst[e]*NHEAD + h], z);
  }
}
__global__ void k_seg_norm(float* __restrict__ attw, const int* __restrict__ edst,
                           const float* __restrict__ nsum){
  int idx = blockIdx.x*blockDim.x + threadIdx.x;
  if (idx < NEDGE*NHEAD){
    int e = idx >> 3, h = idx & 7;
    attw[idx] = attw[idx] / (nsum[edst[e]*NHEAD + h] + EPSF);
  }
}

// ---------------- value messages + scatter: agg[i] += attw * ((xs*gate)@Wv) ----------------
__global__ void k_value(const unsigned short* __restrict__ xn, const float* __restrict__ gate,
                        const float* __restrict__ attw, const unsigned short* __restrict__ Wvp,
                        const int* __restrict__ esrc, const int* __restrict__ edst,
                        float* __restrict__ agg){
  __shared__ int sh_i[4][16];
  __shared__ int sh_e[4][16];
  int wv = threadIdx.x >> 5, lane = threadIdx.x & 31;
  int half = lane >> 4, r = lane & 15;
  int row0 = (blockIdx.x*4 + wv) * 16;
  int gr = row0 + r;
  int e  = gr / NCOEF;
  int lc = gr - e*NCOEF;
  int jN = esrc[e];
  if (lane < 16){ sh_e[wv][r] = e; sh_i[wv][r] = edst[e]; }
  __syncthreads();
  const unsigned short* xrow = xn + ((size_t)jN*NCOEF + lc)*CCH;
  const float* grow = gate + (size_t)e*CCH;
  bf16x16 a[4];
#pragma unroll
  for (int kt=0;kt<4;++kt){
    union { bf16x16 v; unsigned short u[16]; } f;
#pragma unroll
    for (int g=0; g<2; ++g){
      int base = kt*32 + 16*g + 8*half;
#pragma unroll
      for (int q=0;q<8;++q)
        f.u[8*g+q] = f2bf(bf2f(xrow[base+q]) * grow[base+q]);
    }
    a[kt] = f.v;
  }
#pragma unroll
  for (int nt=0;nt<8;++nt){                       // nt == head (V==16)
    f32x8 acc = zero8();
#pragma unroll
    for (int kt=0;kt<4;++kt)
      acc = wmma_bf(a[kt], loadBp(Wvp, 8, kt, nt, lane), acc);
#pragma unroll
    for (int p=0;p<8;++p){
      int m = p + 8*half;
      int em = sh_e[wv][m], im = sh_i[wv][m];
      int lcm = row0 + m - em*NCOEF;
      float val = acc[p] * attw[(size_t)em*NHEAD + nt];
      atomicAdd(&agg[((size_t)im*NCOEF + lcm)*CCH + nt*16 + r], val);
    }
  }
}

// ---------------- fused gated FFN per node ----------------
__global__ void k_ffn(const unsigned short* __restrict__ xn,
                      const unsigned short* __restrict__ W1p,   // packed [128,512]
                      const float* __restrict__ Wg,             // [128,512] f32
                      const unsigned short* __restrict__ W2p,   // packed [512,128]
                      float* __restrict__ x){
  __shared__ __align__(16) unsigned short sh_x[32*128];
  __shared__ float sh_g[FFH];
  __shared__ __align__(16) unsigned short sh_h[2][16*32];
  int node = blockIdx.x, tid = threadIdx.x;   // 64 threads, 2 waves
  for (int idx = tid; idx < 32*128; idx += 64){
    int rowm = idx >> 7, c = idx & 127;
    sh_x[idx] = (rowm < NCOEF) ? xn[((size_t)node*NCOEF + rowm)*CCH + c] : (unsigned short)0;
  }
  __syncthreads();
  for (int cc = tid; cc < FFH; cc += 64){
    float s = 0.f;
    for (int k = 0; k < CCH; ++k) s += bf2f(sh_x[k]) * Wg[(size_t)k*FFH + cc];
    sh_g[cc] = silu_f(s);
  }
  __syncthreads();
  int wv = tid >> 5, lane = tid & 31;
  int half = lane >> 4, r = lane & 15;
  bf16x16 a1[4];
  {
    const unsigned short* rp = sh_x + (16*wv + r)*CCH;
#pragma unroll
    for (int kt=0;kt<4;++kt) a1[kt] = loadA_row(rp, kt*32, half);
  }
  f32x8 acc[8];
#pragma unroll
  for (int nt=0;nt<8;++nt) acc[nt] = zero8();
  for (int t = 0; t < 16; ++t){                 // FF chunks of 32
#pragma unroll
    for (int nn=0;nn<2;++nn){
      f32x8 h8 = zero8();
#pragma unroll
      for (int kt=0;kt<4;++kt)
        h8 = wmma_bf(a1[kt], loadBp(W1p, 32, kt, 2*t + nn, lane), h8);
#pragma unroll
      for (int p=0;p<8;++p){
        int m = p + 8*half, col = 32*t + 16*nn + r;
        sh_h[wv][m*32 + 16*nn + r] = f2bf(h8[p] * sh_g[col]);
      }
    }
    __syncthreads();
    bf16x16 a2;
    {
      const unsigned short* rp = sh_h[wv] + r*32;
      a2 = loadA_row(rp, 0, half);
    }
#pragma unroll
    for (int nt=0;nt<8;++nt)
      acc[nt] = wmma_bf(a2, loadBp(W2p, 8, t, nt, lane), acc[nt]);
    __syncthreads();
  }
#pragma unroll
  for (int nt=0;nt<8;++nt){
#pragma unroll
    for (int p=0;p<8;++p){
      int m = p + 8*half, row = 16*wv + m;
      if (row < NCOEF){
        size_t o = ((size_t)node*NCOEF + row)*CCH + nt*16 + r;
        x[o] += acc[nt][p];
      }
    }
  }
}

// ---------------- host driver ----------------
extern "C" void kernel_launch(void* const* d_in, const int* in_sizes, int n_in,
                              void* d_out, int out_size, void* d_ws, size_t ws_size,
                              hipStream_t stream){
  (void)in_sizes; (void)n_in; (void)out_size; (void)ws_size;
  const float* pos       = (const float*)d_in[0];
  const float* emb       = (const float*)d_in[1];
  const float* W_rad1    = (const float*)d_in[2];
  const float* b_rad1    = (const float*)d_in[3];
  const float* W_rad2    = (const float*)d_in[4];
  const float* b_rad2    = (const float*)d_in[5];
  const float* W_alpha   = (const float*)d_in[6];
  const float* alpha_vec = (const float*)d_in[7];
  const float* W_gr      = (const float*)d_in[8];
  const float* W_v       = (const float*)d_in[9];
  const float* W_o       = (const float*)d_in[10];
  const float* W1        = (const float*)d_in[11];
  const float* Wg        = (const float*)d_in[12];
  const float* W2        = (const float*)d_in[13];
  const float* s_attn    = (const float*)d_in[14];
  const float* s_ffn     = (const float*)d_in[15];
  const int*   anum      = (const int*)d_in[16];
  const int*   eidx      = (const int*)d_in[17];
  float* x = (float*)d_out;
  const int* esrc = eidx;           // j
  const int* edst = eidx + NEDGE;   // i

  char* ws = (char*)d_ws;
  size_t off = 0;
  auto alloc = [&](size_t bytes)->char*{
    char* p = ws + off;
    off = (off + bytes + 255) & ~(size_t)255;
    return p;
  };
  float*          dist  = (float*)alloc((size_t)NEDGE*4);
  unsigned short* wr1b  = (unsigned short*)alloc((size_t)NLAY*NBPAD*CCH*2);
  unsigned short* wr2b  = (unsigned short*)alloc((size_t)NLAY*CCH*CCH*2);
  unsigned short* wab   = (unsigned short*)alloc((size_t)NLAY*2*CCH*2*CCH*2);
  unsigned short* wgrb  = (unsigned short*)alloc((size_t)NLAY*CCH*CCH*2);
  unsigned short* wvb   = (unsigned short*)alloc((size_t)NLAY*CCH*CCH*2);
  unsigned short* wob   = (unsigned short*)alloc((size_t)NLAY*CCH*CCH*2);
  unsigned short* w1b   = (unsigned short*)alloc((size_t)NLAY*CCH*FFH*2);
  unsigned short* w2b   = (unsigned short*)alloc((size_t)NLAY*FFH*CCH*2);
  unsigned short* e1b   = (unsigned short*)alloc((size_t)NEDGE*CCH*2);
  unsigned short* eb    = (unsigned short*)alloc((size_t)NEDGE*CCH*2);
  float*          gate  = (float*)alloc((size_t)NEDGE*CCH*4);
  unsigned short* xnb   = (unsigned short*)alloc((size_t)NATOM*NCOEF*CCH*2);
  float*          logit = (float*)alloc((size_t)NEDGE*NHEAD*4);
  float*          attw  = (float*)alloc((size_t)NEDGE*NHEAD*4);
  unsigned*       nmax  = (unsigned*)alloc((size_t)NATOM*NHEAD*4);
  float*          nsum  = (float*)alloc((size_t)NATOM*NHEAD*4);
  float*          agg   = (float*)alloc((size_t)NATOM*NCOEF*CCH*4);

  // pack all weights into B-fragment layout (per layer)
#define PACKB(dst, src, K, N, Kp) \
  k_pack_b<<<((((Kp)>>5)*((N)>>4)*256 + 255)/256), 256, 0, stream>>>(src, dst, K, N, Kp)
  for (int l = 0; l < NLAY; ++l){
    PACKB(wr1b + (size_t)l*NBPAD*CCH,     W_rad1  + (size_t)l*NBASIS*CCH, NBASIS, CCH, NBPAD);
    PACKB(wr2b + (size_t)l*CCH*CCH,       W_rad2  + (size_t)l*CCH*CCH,    CCH, CCH, CCH);
    PACKB(wab  + (size_t)l*2*CCH*2*CCH,   W_alpha + (size_t)l*2*CCH*2*CCH, 2*CCH, 2*CCH, 2*CCH);
    PACKB(wgrb + (size_t)l*CCH*CCH,       W_gr    + (size_t)l*CCH*CCH,    CCH, CCH, CCH);
    PACKB(wvb  + (size_t)l*CCH*CCH,       W_v     + (size_t)l*CCH*CCH,    CCH, CCH, CCH);
    PACKB(wob  + (size_t)l*CCH*CCH,       W_o     + (size_t)l*CCH*CCH,    CCH, CCH, CCH);
    PACKB(w1b  + (size_t)l*CCH*FFH,       W1      + (size_t)l*CCH*FFH,    CCH, FFH, CCH);
    PACKB(w2b  + (size_t)l*FFH*CCH,       W2      + (size_t)l*FFH*CCH,    FFH, CCH, FFH);
  }
#undef PACKB

  k_dist<<<(NEDGE+255)/256, 256, 0, stream>>>(pos, esrc, edst, dist);
  k_init_x<<<(NATOM*NCOEF*CCH+255)/256, 256, 0, stream>>>(emb, anum, x);

  for (int l = 0; l < NLAY; ++l){
    const unsigned short* wr1 = wr1b + (size_t)l*NBPAD*CCH;
    const unsigned short* wr2 = wr2b + (size_t)l*CCH*CCH;
    const unsigned short* wa  = wab  + (size_t)l*2*CCH*2*CCH;
    const unsigned short* wgr = wgrb + (size_t)l*CCH*CCH;
    const unsigned short* wv  = wvb  + (size_t)l*CCH*CCH;
    const unsigned short* wo  = wob  + (size_t)l*CCH*CCH;
    const unsigned short* w1  = w1b  + (size_t)l*CCH*FFH;
    const unsigned short* w2  = w2b  + (size_t)l*FFH*CCH;

    // radial MLP + gate
    k_rad1<<<NEDGE/64, 128, 0, stream>>>(dist, wr1, b_rad1 + l*CCH, e1b);
    k_gemm128<<<NEDGE/64, 128, 0, stream>>>(e1b, nullptr, wr2, b_rad2 + l*CCH,
                                            eb, nullptr, NEDGE, 0, 0);
    k_gemm128<<<NEDGE/64, 128, 0, stream>>>(eb, nullptr, wgr, nullptr,
                                            nullptr, gate, NEDGE, 1, 0);
    // attention
    k_rms<<<NATOM, 256, 0, stream>>>(x, s_attn + l*CCH, xnb);
    k_alpha<<<NEDGE/64, 128, 0, stream>>>(xnb, wa, alpha_vec + l*NHEAD*ACH, esrc, edst, logit);
    k_zero_u<<<(NATOM*NHEAD+255)/256, 256, 0, stream>>>(nmax, NATOM*NHEAD);
    k_zero_f<<<(NATOM*NHEAD+255)/256, 256, 0, stream>>>(nsum, NATOM*NHEAD);
    k_seg_max<<<(NEDGE*NHEAD)/256, 256, 0, stream>>>(logit, edst, nmax);
    k_seg_exp<<<(NEDGE*NHEAD)/256, 256, 0, stream>>>(logit, edst, nmax, nsum, attw);
    k_seg_norm<<<(NEDGE*NHEAD)/256, 256, 0, stream>>>(attw, edst, nsum);
    // messages + aggregation + output projection
    k_zero_f<<<(ROWS_N*CCH+255)/256, 256, 0, stream>>>(agg, ROWS_N*CCH);
    k_value<<<ROWS_E/64, 128, 0, stream>>>(xnb, gate, attw, wv, esrc, edst, agg);
    k_gemm128<<<ROWS_N/64, 128, 0, stream>>>(nullptr, agg, wo, nullptr,
                                             nullptr, x, ROWS_N, 0, 1);
    // gated FFN
    k_rms<<<NATOM, 256, 0, stream>>>(x, s_ffn + l*CCH, xnb);
    k_ffn<<<NATOM, 64, 0, stream>>>(xnb, w1, Wg + (size_t)l*CCH*FFH, w2, x);
  }
}
